// Head_9345848836150
// MI455X (gfx1250) — matrix operand order repaired and checked by reading, compiled, not run on hardware
//
#include <hip/hip_runtime.h>

// ---------------------------------------------------------------------------
// Problem dims (fixed by the reference): B=64, T=1024, D=512, H=512
// ---------------------------------------------------------------------------
#define B_ 64
#define T_ 1024
#define D_ 512
#define H_ 512
#define WAVES 2
#define NEGINF (-1e30f)

typedef __attribute__((ext_vector_type(16))) __bf16 v16bf;
typedef __attribute__((ext_vector_type(8)))  float  v8f;
typedef __attribute__((ext_vector_type(4)))  unsigned int u32x4;
typedef __attribute__((ext_vector_type(8)))  int          i32x8;
typedef __attribute__((ext_vector_type(4)))  int          i32x4;

// TDM availability (device pass only; host pass sees x86 builtins -> 0)
#if __has_builtin(__builtin_amdgcn_tensor_load_to_lds) && \
    __has_builtin(__builtin_amdgcn_s_wait_tensorcnt)
#define HAVE_TDM 1
#else
#define HAVE_TDM 0
#endif

// LDS layout for the attention kernel (bytes)
#define SLOG_BYTES (WAVES * 16 * T_ * 4)            // f32 logits   131072
#define SABF_BYTES (WAVES * 16 * T_ * 2)            // bf16 A       65536
#define KPAN_BYTES (H_ * 16 * 2)                    // one K^T panel 16384
#define LDS_TOTAL  (SLOG_BYTES + SABF_BYTES + WAVES * 2 * KPAN_BYTES)  // 262144

union Frag16 {
  v16bf v;
  unsigned short h[16];
  unsigned int d[8];
  uint4 q[2];
};

// f32 -> bf16 round-to-nearest-even
__device__ __forceinline__ unsigned short f2bf(float f) {
  unsigned int u = __builtin_bit_cast(unsigned int, f);
  u += 0x7FFFu + ((u >> 16) & 1u);
  return (unsigned short)(u >> 16);
}

// ---------------------------------------------------------------------------
// Tensor Data Mover: load a [H_ x 16] bf16 panel (row stride T_ elements)
// into LDS at byte offset lds_off.  D# packing per CDNA5 ISA §8.3 / §8.4.
// ---------------------------------------------------------------------------
#if HAVE_TDM
__device__ __forceinline__ void tdm_load_kpanel(const unsigned short* gsrc,
                                                unsigned int lds_off) {
  const unsigned long long ga = (unsigned long long)(const void*)gsrc;
  u32x4 g0;
  g0.x = 1u;                                          // count=1, user D#
  g0.y = lds_off;                                     // lds_addr (bytes)
  g0.z = (unsigned int)ga;                            // global_addr[31:0]
  g0.w = (unsigned int)((ga >> 32) & 0x1FFFFFFull)    // global_addr[56:32]
         | (2u << 30);                                // type = 2 ("image")
  i32x8 g1;
  g1[0] = 1 << 16;                                    // data_size = 2 bytes
  g1[1] = (int)(16u << 16);                           // tensor_dim0 = 16 (lo)
  g1[2] = (int)((unsigned)H_ << 16);                  // tensor_dim1 = 512 (lo)
  g1[3] = (int)(16u << 16);                           // tile_dim0 = 16
  g1[4] = H_;                                         // tile_dim1 = 512, tile_dim2 = 0
  g1[5] = T_;                                         // tensor_dim0_stride = 1024
  g1[6] = 0;
  g1[7] = 0;
  i32x4 z4 = {0, 0, 0, 0};
#if defined(__clang_major__) && __clang_major__ >= 23
  i32x8 z8 = {0, 0, 0, 0, 0, 0, 0, 0};
  __builtin_amdgcn_tensor_load_to_lds(g0, g1, z4, z4, z8, 0);
#else
  __builtin_amdgcn_tensor_load_to_lds(g0, g1, z4, z4, 0);
#endif
}
#endif

// ---------------------------------------------------------------------------
// Kernel 1: projection GEMM, one wave per 16x16 output tile.
//   TRANS=false : Dst[row, col]      (bf16 [B*T, H])   -> Q, V
//   TRANS=true  : Dst[b, col, t]     (bf16 [B, H, T])  -> K transposed
// ---------------------------------------------------------------------------
template <bool TRANS>
__global__ void __launch_bounds__(256)
proj_kernel(const float* __restrict__ In, const float* __restrict__ W,
            const float* __restrict__ bias, unsigned short* __restrict__ Dst) {
  const int lane = threadIdx.x & 31;
  const int wave = threadIdx.x >> 5;
  const int rowbase = blockIdx.x * 128 + wave * 16;   // row in [0, B*T)
  const int colbase = blockIdx.y * 16;                // col in [0, H)
  const int m  = lane & 15;
  const int kh = lane >> 4;

  v8f acc = {};
  const float* arow = In + (size_t)(rowbase + m) * D_;
  for (int k0 = 0; k0 < D_; k0 += 32) {
    Frag16 a, b;
    // A fragment: K = k0 + kh*8 + {0..7} and {16..23}  -> 4x float4
    const float* ap = arow + k0 + kh * 8;
    const float4 f0 = *(const float4*)(ap);
    const float4 f1 = *(const float4*)(ap + 4);
    const float4 f2 = *(const float4*)(ap + 16);
    const float4 f3 = *(const float4*)(ap + 20);
    a.h[0]  = f2bf(f0.x); a.h[1]  = f2bf(f0.y); a.h[2]  = f2bf(f0.z); a.h[3]  = f2bf(f0.w);
    a.h[4]  = f2bf(f1.x); a.h[5]  = f2bf(f1.y); a.h[6]  = f2bf(f1.z); a.h[7]  = f2bf(f1.w);
    a.h[8]  = f2bf(f2.x); a.h[9]  = f2bf(f2.y); a.h[10] = f2bf(f2.z); a.h[11] = f2bf(f2.w);
    a.h[12] = f2bf(f3.x); a.h[13] = f2bf(f3.y); a.h[14] = f2bf(f3.z); a.h[15] = f2bf(f3.w);
    // B fragment: N = lane&15, K = k0 + kh*8 + {0..7} and {16..23}
    const float* wp = W + (size_t)(k0 + kh * 8) * H_ + colbase + m;
#pragma unroll
    for (int s = 0; s < 16; ++s)
      b.h[s] = f2bf(wp[(size_t)((s & 7) + (s >> 3) * 16) * H_]);

    acc = __builtin_amdgcn_wmma_f32_16x16x32_bf16(false, a.v, false, b.v,
                                                  (short)0, acc, false, false);
  }

  // C layout: N = lane&15, M = r + 8*kh
  const int n = lane & 15;
  const float bb = bias[colbase + n];
  if (TRANS) {
    const int bidx = rowbase >> 10;                    // rowbase / T_
    const int t0   = (rowbase & (T_ - 1)) + kh * 8;    // 8 consecutive t's
    Frag16 o;
#pragma unroll
    for (int r = 0; r < 8; ++r) o.h[r] = f2bf(acc[r] + bb);
    *(uint4*)(Dst + ((size_t)bidx * H_ + colbase + n) * T_ + t0) = o.q[0];
  } else {
    unsigned short* p = Dst + (size_t)(rowbase + kh * 8) * H_ + colbase + n;
#pragma unroll
    for (int r = 0; r < 8; ++r) p[(size_t)r * H_] = f2bf(acc[r] + bb);
  }
}

// ---------------------------------------------------------------------------
// Kernel 2: fused scores + mask + softmax + output GEMM.
// One wave owns 16 query rows.  K^T panels are staged through LDS by the
// Tensor Data Mover (double buffered), scores live in a wave-private LDS slab.
// ---------------------------------------------------------------------------
__global__ void __launch_bounds__(WAVES * 32)
attn_kernel(const unsigned short* __restrict__ Qbf,
            const unsigned short* __restrict__ Ktbf,
            const unsigned short* __restrict__ Vbf,
            const float* __restrict__ dag,
            float* __restrict__ Out) {
  extern __shared__ char smem[];
  const int lane  = threadIdx.x & 31;
  const int wave  = threadIdx.x >> 5;
  const int b     = blockIdx.x;
  const int tbase = blockIdx.y * (16 * WAVES) + wave * 16;

  float* slog = (float*)smem + (size_t)wave * 16 * T_;
  unsigned short* sabf =
      (unsigned short*)(smem + SLOG_BYTES) + (size_t)wave * 16 * T_;

  const int m  = lane & 15;
  const int kh = lane >> 4;

  // ---- preload Q fragments (16 K-steps covering H=512) ----
  Frag16 qf[16];
  {
    const unsigned short* qrow = Qbf + (size_t)(b * T_ + tbase + m) * H_;
#pragma unroll
    for (int kf = 0; kf < 16; ++kf) {
      const unsigned short* qp = qrow + kf * 32 + kh * 8;
      qf[kf].q[0] = *(const uint4*)(qp);
      qf[kf].q[1] = *(const uint4*)(qp + 16);
    }
  }

  // ---- scores S = Q K^T, masked, into LDS ----
  const float scale = 0.044194173824159216f;  // 1/sqrt(512)
#if HAVE_TDM
  const unsigned short* ktbase = Ktbf + (size_t)b * H_ * T_;
  const unsigned int kpan_off = SLOG_BYTES + SABF_BYTES + wave * 2 * KPAN_BYTES;
  const unsigned short* kpan = (const unsigned short*)(smem + kpan_off);
  tdm_load_kpanel(ktbase, kpan_off);                 // panel for s-tile 0
#else
  const unsigned short* ktlane = Ktbf + ((size_t)b * H_ + lane) * T_;
#endif

  for (int st = 0; st < T_ / 16; ++st) {
    const int sbase = st * 16;
#if HAVE_TDM
    if (st + 1 < T_ / 16) {
      tdm_load_kpanel(ktbase + sbase + 16,
                      kpan_off + (unsigned)((st + 1) & 1) * KPAN_BYTES);
      __builtin_amdgcn_s_wait_tensorcnt(1);          // current panel ready
    } else {
      __builtin_amdgcn_s_wait_tensorcnt(0);
    }
    const unsigned short* pan = kpan + (size_t)(st & 1) * (KPAN_BYTES / 2);
#endif
    v8f c = {};
#pragma unroll
    for (int kf = 0; kf < 16; ++kf) {
      Frag16 bfr;
#if HAVE_TDM
      const unsigned short* p = pan + (size_t)(kf * 32 + kh * 8) * 16 + m;
#pragma unroll
      for (int s2 = 0; s2 < 16; ++s2)
        bfr.h[s2] = p[((s2 & 7) + (s2 >> 3) * 16) * 16];
#else
      const unsigned short* p =
          Ktbf + ((size_t)b * H_ + kf * 32 + kh * 8) * T_ + sbase + m;
#pragma unroll
      for (int s2 = 0; s2 < 16; ++s2)
        bfr.h[s2] = p[(size_t)((s2 & 7) + (s2 >> 3) * 16) * T_];
#endif
      c = __builtin_amdgcn_wmma_f32_16x16x32_bf16(false, qf[kf].v, false,
                                                  bfr.v, (short)0, c,
                                                  false, false);
    }
    const int n = lane & 15;
    const int s = sbase + n;
#pragma unroll
    for (int r = 0; r < 8; ++r) {
      const int mm = r + kh * 8;
      const float g  = dag[(size_t)s * T_ + (tbase + mm)];  // dag.T[t][s]
      const float sp = c[r] * scale * g;
      slog[mm * T_ + s] = (sp == 0.0f) ? NEGINF : sp;
    }
  }

  // ---- softmax: 2 lanes per row (wave-private LDS; DS ops are in-order) ----
  const int rrow = lane >> 1;
  const int half = lane & 1;
  float* lr = slog + rrow * T_ + half * (T_ / 2);
  float rmax = -3.0e38f;
  for (int j = 0; j < T_ / 2; ++j) rmax = fmaxf(rmax, lr[j]);
  rmax = fmaxf(rmax, __shfl_xor(rmax, 1));
  const bool dead = (rmax < -1.0e29f);

  float sum = 0.0f;
  for (int j = 0; j < T_ / 2; ++j) {
    const float e = __expf(lr[j] - rmax);
    lr[j] = e;
    sum += e;
  }
  sum += __shfl_xor(sum, 1);
  const float inv = dead ? 0.0f : (1.0f / sum);

  unsigned short* aw = sabf + rrow * T_ + half * (T_ / 2);
  for (int j = 0; j < T_ / 2; ++j) aw[j] = f2bf(lr[j] * inv);

  // ---- O = A V, 64 output columns at a time ----
  for (int hc = 0; hc < H_ / 64; ++hc) {
    v8f oc0 = {}, oc1 = {}, oc2 = {}, oc3 = {};
    for (int k0 = 0; k0 < T_; k0 += 32) {
      Frag16 af;
      const unsigned short* ap = sabf + m * T_ + k0 + kh * 8;
      af.q[0] = *(const uint4*)(ap);
      af.q[1] = *(const uint4*)(ap + 16);

      const unsigned short* vrow =
          Vbf + (size_t)(b * T_ + k0 + kh * 8) * H_ + hc * 64 + m;
      __builtin_prefetch(vrow + 32 * H_, 0, 1);  // global_prefetch_b8
      Frag16 b0, b1, b2, b3;
#pragma unroll
      for (int s2 = 0; s2 < 16; ++s2) {
        const size_t off = (size_t)((s2 & 7) + (s2 >> 3) * 16) * H_;
        b0.h[s2] = vrow[off];
        b1.h[s2] = vrow[off + 16];
        b2.h[s2] = vrow[off + 32];
        b3.h[s2] = vrow[off + 48];
      }

      oc0 = __builtin_amdgcn_wmma_f32_16x16x32_bf16(false, af.v, false, b0.v,
                                                    (short)0, oc0, false, false);
      oc1 = __builtin_amdgcn_wmma_f32_16x16x32_bf16(false, af.v, false, b1.v,
                                                    (short)0, oc1, false, false);
      oc2 = __builtin_amdgcn_wmma_f32_16x16x32_bf16(false, af.v, false, b2.v,
                                                    (short)0, oc2, false, false);
      oc3 = __builtin_amdgcn_wmma_f32_16x16x32_bf16(false, af.v, false, b3.v,
                                                    (short)0, oc3, false, false);
    }
    const int n = lane & 15;
    float* orow = Out + (size_t)(b * T_ + tbase) * H_ + hc * 64 + n;
#pragma unroll
    for (int r = 0; r < 8; ++r) {
      const int mm = r + kh * 8;
      orow[(size_t)mm * H_ + 0]  = oc0[r];
      orow[(size_t)mm * H_ + 16] = oc1[r];
      orow[(size_t)mm * H_ + 32] = oc2[r];
      orow[(size_t)mm * H_ + 48] = oc3[r];
    }
  }
}

// ---------------------------------------------------------------------------
// Host launch
// ---------------------------------------------------------------------------
extern "C" void kernel_launch(void* const* d_in, const int* in_sizes, int n_in,
                              void* d_out, int out_size, void* d_ws,
                              size_t ws_size, hipStream_t stream) {
  (void)in_sizes; (void)n_in; (void)out_size; (void)ws_size;

  const float* X   = (const float*)d_in[0];
  const float* Y   = (const float*)d_in[1];
  const float* dag = (const float*)d_in[2];
  const float* Wk  = (const float*)d_in[3];
  const float* bk  = (const float*)d_in[4];
  const float* Wq  = (const float*)d_in[5];
  const float* bq  = (const float*)d_in[6];
  const float* Wv  = (const float*)d_in[7];
  const float* bv  = (const float*)d_in[8];
  float* Out = (float*)d_out;

  const size_t elems = (size_t)B_ * T_ * H_;
  unsigned short* Qbf  = (unsigned short*)d_ws;
  unsigned short* Ktbf = Qbf + elems;
  unsigned short* Vbf  = Ktbf + elems;

  dim3 g1(B_ * T_ / 128, H_ / 16);
  proj_kernel<false><<<g1, 256, 0, stream>>>(Y, Wq, bq, Qbf);
  proj_kernel<true ><<<g1, 256, 0, stream>>>(X, Wk, bk, Ktbf);
  proj_kernel<false><<<g1, 256, 0, stream>>>(X, Wv, bv, Vbf);

  dim3 g2(B_, T_ / (16 * WAVES));
  attn_kernel<<<g2, WAVES * 32, LDS_TOTAL, stream>>>(Qbf, Ktbf, Vbf, dag, Out);
}
